// DisAttLayer_11458972746001
// MI455X (gfx1250) — matrix operand order repaired and checked by reading, not compiled, not run on hardware
//
#include <hip/hip_runtime.h>

typedef __attribute__((ext_vector_type(16))) _Float16 v16h;
typedef __attribute__((ext_vector_type(8)))  float    v8f;

// Problem constants (reference): B=4, S=256, H=8, hidden 96->32->16->1
#define S 256
#define NH 8
#define NB 4
#define PADK 36                       // LDS row stride in floats (bank-conflict-free)

// ---------------------------------------------------------------------------
// Precompute P1[h,f,k] = sum_d e_pos[f,h,d] * w1[d,k,h]   (f<512, k<32)
// Head-major layout so the per-head slice is one contiguous 64KB block.
// ---------------------------------------------------------------------------
__global__ __launch_bounds__(256) void p1_kernel(const float* __restrict__ e_pos,
                                                 const float* __restrict__ w1,
                                                 float* __restrict__ P1) {
  int tid = blockIdx.x * blockDim.x + threadIdx.x;   // 512*8*32 = 131072
  int k = tid & 31;
  int h = (tid >> 5) & 7;
  int f = tid >> 8;
  const float* pe = e_pos + (f * NH + h) * 32;
  float acc = 0.f;
#pragma unroll
  for (int d = 0; d < 32; ++d)
    acc = fmaf(pe[d], w1[(d * 32 + k) * NH + h], acc);
  P1[((h << 9) + f) * 32 + k] = acc;
}

// ---------------------------------------------------------------------------
// Precompute (head-major: [b][h][m][k], contiguous 32KB per (b,h) slice)
//   R1[b,h,m,k] = sum_t bi[t]*w1[32+t,k,h] + ci[t]*w1[64+t,k,h]   (row terms)
//   C1[b,h,n,k] = sum_t bj[t]*w1[48+t,k,h] + cj[t]*w1[80+t,k,h]   (col terms)
// ---------------------------------------------------------------------------
__global__ __launch_bounds__(256) void rc_kernel(const int* __restrict__ b_seq,
                                                 const int* __restrict__ c_seq,
                                                 const float* __restrict__ e_bi,
                                                 const float* __restrict__ e_bj,
                                                 const float* __restrict__ e_ci,
                                                 const float* __restrict__ e_cj,
                                                 const float* __restrict__ w1,
                                                 float* __restrict__ R1,
                                                 float* __restrict__ C1) {
  int tid = blockIdx.x * blockDim.x + threadIdx.x;   // 4*256*8*32 = 262144
  int k = tid & 31;
  int h = (tid >> 5) & 7;
  int m = (tid >> 8) & (S - 1);
  int b = tid >> 16;
  int bs = b_seq[b * S + m];
  int cs = c_seq[b * S + m];
  const float* bi = e_bi + (bs * NH + h) * 16;
  const float* bj = e_bj + (bs * NH + h) * 16;
  const float* ci = e_ci + (cs * NH + h) * 16;
  const float* cj = e_cj + (cs * NH + h) * 16;
  float r = 0.f, c = 0.f;
#pragma unroll
  for (int t = 0; t < 16; ++t) {
    r = fmaf(bi[t], w1[((32 + t) * 32 + k) * NH + h],
        fmaf(ci[t], w1[((64 + t) * 32 + k) * NH + h], r));
    c = fmaf(bj[t], w1[((48 + t) * 32 + k) * NH + h],
        fmaf(cj[t], w1[((80 + t) * 32 + k) * NH + h], c));
  }
  int o = ((b * NH + h) * S + m) * 32 + k;
  R1[o] = r;
  C1[o] = c;
}

// ---------------------------------------------------------------------------
// CDNA5 async global->LDS copy (16B per lane) + wait, per ISA 08_async_tensor
// ---------------------------------------------------------------------------
__device__ inline void async_copy16(unsigned lds_off, const void* gptr) {
  asm volatile("global_load_async_to_lds_b128 %0, %1, off"
               :: "v"(lds_off), "v"((unsigned long long)(uintptr_t)gptr)
               : "memory");
}
__device__ inline void wait_async0() {
  asm volatile("s_wait_asynccnt 0x0" ::: "memory");
}

__device__ inline void load16f(const float* __restrict__ p, float* dst) {
  float4 a = ((const float4*)p)[0];
  float4 b = ((const float4*)p)[1];
  float4 c = ((const float4*)p)[2];
  float4 d = ((const float4*)p)[3];
  dst[0] = a.x; dst[1] = a.y; dst[2]  = a.z; dst[3]  = a.w;
  dst[4] = b.x; dst[5] = b.y; dst[6]  = b.z; dst[7]  = b.w;
  dst[8] = c.x; dst[9] = c.y; dst[10] = c.z; dst[11] = c.w;
  dst[12] = d.x; dst[13] = d.y; dst[14] = d.z; dst[15] = d.w;
}

// Build one B-matrix tile: B[k, cell] = relu(P1 + R1 + C1) in f16 (h1 = layer1)
__device__ inline v16h build_B(const float* __restrict__ P1s,
                               const float* __restrict__ C1s,
                               const float* rr, int m, int n0, int l, int k0) {
  int cell = n0 + l;
  int f    = m - cell + S;        // relative-position row, in [1, 511]
  float pv[16], cv[16];
  load16f(P1s + f * PADK + k0, pv);      // ds_load_b128 x4, conflict-free
  load16f(C1s + cell * PADK + k0, cv);
  v16h Bm;
#pragma unroll
  for (int j = 0; j < 16; ++j)
    Bm[j] = (_Float16)fmaxf(pv[j] + rr[j] + cv[j], 0.f);
  return Bm;
}

// score partial: sum_l relu(D[l,cell]) * w3[l] over this lane-half's 8 rows
__device__ inline float dotrelu(v8f D, const float* w3v) {
  float part = 0.f;
#pragma unroll
  for (int v = 0; v < 8; ++v)
    part = fmaf(fmaxf(D[v], 0.f), w3v[v], part);
  return part;
}

// ---------------------------------------------------------------------------
// Main kernel: block = 8 waves, fixed (b,h), m = mg*8 + waveId.
// Async-stage P1 h-slice (512x32) and C1 (b,h)-slice (256x32) into LDS
// (rows padded to 36 floats -> uniform 2-way LDS banking), then per wave:
// 8 pipelined tile-pairs, one v_wmma_f32_16x16x32_f16 per 16 cells. The
// next pair's B-build fills the WMMA->VALU hazard slots; both lane halves
// store (tile-even / tile-odd) so every global_store_b32 is unmasked.
// ---------------------------------------------------------------------------
__global__ __launch_bounds__(256) void score_kernel(const float* __restrict__ P1,
                                                    const float* __restrict__ R1,
                                                    const float* __restrict__ C1,
                                                    const float* __restrict__ w2,
                                                    const float* __restrict__ w3,
                                                    float* __restrict__ out) {
  extern __shared__ float smem[];     // 512*36 + 256*36 = 27648 floats (108KB)
  float* P1s = smem;
  float* C1s = smem + 512 * PADK;

  int blk = blockIdx.x;               // 1024 blocks
  int mg = blk & 31;
  int h  = (blk >> 5) & 7;
  int b  = blk >> 8;
  int m  = mg * 8 + (threadIdx.x >> 5);

  // ---- async stage tables into LDS (flat 16B chunks, re-padded rows) ----
  {
    const char* P1g = (const char*)(P1 + (h << 9) * 32);           // 64KB slice
    const char* C1g = (const char*)(C1 + ((b * NH + h) * S) * 32); // 32KB slice
    unsigned p1b = (unsigned)(uintptr_t)P1s;
    unsigned c1b = (unsigned)(uintptr_t)C1s;
    unsigned t = threadIdx.x;
#pragma unroll
    for (int i = 0; i < 16; ++i) {                 // 4096 chunks of 16B
      unsigned chunk = t + i * 256;
      unsigned row = chunk >> 3, q = chunk & 7;
      async_copy16(p1b + row * (PADK * 4) + q * 16, P1g + chunk * 16);
    }
#pragma unroll
    for (int i = 0; i < 8; ++i) {                  // 2048 chunks of 16B
      unsigned chunk = t + i * 256;
      unsigned row = chunk >> 3, q = chunk & 7;
      async_copy16(c1b + row * (PADK * 4) + q * 16, C1g + chunk * 16);
    }
    wait_async0();
    __syncthreads();
  }

  int lane = threadIdx.x & 31;
  bool lo = lane < 16;
  int  l  = lane & 15;        // A: row l ; B/D: column (cell) index
  int  k0 = lo ? 0 : 16;      // B-matrix: contiguous 16-K run per lane half

  // A = w2^T : lane half selects K-subsets {0..15} / {16..31}
  v16h A;
#pragma unroll
  for (int j = 0; j < 16; ++j) {
    int k = j + (lo ? 0 : 16);
    A[j] = (_Float16)w2[(k * 16 + l) * NH + h];
  }
  // w3 slice for D rows owned by this lane half (M=0..7 or M=8..15)
  float w3v[8];
#pragma unroll
  for (int v = 0; v < 8; ++v)
    w3v[v] = w3[((lo ? 0 : 8) + v) * NH + h];

  // Row term: constant over the whole wave (fixed b,h,m); one global read
  float rr[16];
  load16f(R1 + ((b * NH + h) * S + m) * 32 + k0, rr);

  const int outbase = ((b * NH + h) * S + m) * S;

  // ---- software-pipelined main loop: 8 tile-pairs, fully unrolled ----
  v16h B0 = build_B(P1s, C1s, rr, m, 0, l, k0);
  v16h B1 = build_B(P1s, C1s, rr, m, 16, l, k0);
#pragma unroll
  for (int it = 0; it < 8; ++it) {
    v8f Cz = {};
    v8f D0 = __builtin_amdgcn_wmma_f32_16x16x32_f16(
        false, A, false, B0, (short)0, Cz, false, false);
    v16h Bn0 = B0, Bn1 = B1;
    if (it < 7)  // build next pair between WMMA issue and D consumption
      Bn0 = build_B(P1s, C1s, rr, m, it * 32 + 32, l, k0);
    float p0 = dotrelu(D0, w3v);

    v8f D1 = __builtin_amdgcn_wmma_f32_16x16x32_f16(
        false, A, false, B1, (short)0, Cz, false, false);
    if (it < 7)
      Bn1 = build_B(P1s, C1s, rr, m, it * 32 + 48, l, k0);
    float p1 = dotrelu(D1, w3v);

    float t0 = p0 + __shfl_xor(p0, 16, 32);   // all lanes hold tile-even score
    float t1 = p1 + __shfl_xor(p1, 16, 32);   // all lanes hold tile-odd score
    out[outbase + it * 32 + lane] = lo ? t0 : t1;  // unmasked coalesced store

    B0 = Bn0; B1 = Bn1;
  }
}

// ---------------------------------------------------------------------------
extern "C" void kernel_launch(void* const* d_in, const int* in_sizes, int n_in,
                              void* d_out, int out_size, void* d_ws, size_t ws_size,
                              hipStream_t stream) {
  (void)in_sizes; (void)n_in; (void)out_size; (void)ws_size;
  const int*   b_seq = (const int*)d_in[0];
  const int*   c_seq = (const int*)d_in[1];
  const float* e_pos = (const float*)d_in[2];
  const float* e_bi  = (const float*)d_in[3];
  const float* e_bj  = (const float*)d_in[4];
  const float* e_ci  = (const float*)d_in[5];
  const float* e_cj  = (const float*)d_in[6];
  const float* w1    = (const float*)d_in[7];
  const float* w2    = (const float*)d_in[8];
  const float* w3    = (const float*)d_in[9];

  float* P1 = (float*)d_ws;                 // 8*512*32   = 131072 f32 (512 KB)
  float* R1 = P1 + NH * 512 * 32;           // 4*8*256*32 = 262144 f32 (1 MB)
  float* C1 = R1 + NB * NH * S * 32;        // 262144 f32 (1 MB)

  p1_kernel<<<512, 256, 0, stream>>>(e_pos, w1, P1);
  rc_kernel<<<1024, 256, 0, stream>>>(b_seq, c_seq, e_bi, e_bj, e_ci, e_cj, w1, R1, C1);
  size_t lds_bytes = (512 + 256) * PADK * sizeof(float);   // 110592 B
  score_kernel<<<1024, 256, lds_bytes, stream>>>(P1, R1, C1, w2, w3, (float*)d_out);
}
